// MSA_46033459478834
// MI455X (gfx1250) — matrix operand order repaired and checked by reading, not compiled, not run on hardware
//
#include <hip/hip_runtime.h>
#include <hip/hip_bf16.h>
#include <cstdint>

// ---------------------------------------------------------------------------
// Problem sizes
// ---------------------------------------------------------------------------
#define BATCH 16
#define SEQ   512
#define EMB   768
#define HEADS 12
#define HDIM  64
#define NLBL  100
#define LPAD  128
#define ROWS  (BATCH * SEQ)          // 8192
#define NEGB  (-1e9f)

typedef __attribute__((ext_vector_type(8)))  _Float16 v8h;
typedef __attribute__((ext_vector_type(16))) _Float16 v16h;
typedef __attribute__((ext_vector_type(8)))  float    v8f;

// ---------------------------------------------------------------------------
// WMMA helpers (CDNA5: V_WMMA_F32_16X16X32_F16, wave32)
// ---------------------------------------------------------------------------
__device__ __forceinline__ v8f wmma_f16(v16h a, v16h b, v8f c) {
    // (neg_a, A, neg_b, B, c_mod, C, reuse_a, reuse_b)
    return __builtin_amdgcn_wmma_f32_16x16x32_f16(false, a, false, b,
                                                  (short)0, c, false, false);
}

__device__ __forceinline__ v16h make_v16(v8h lo, v8h hi) {
    v16h r;
#pragma unroll
    for (int i = 0; i < 8; ++i) { r[i] = lo[i]; r[i + 8] = hi[i]; }
    return r;
}

// A fragment (16x32 f16, M x K). ISA layout: lanes 0-15 = rows, h0..h7 -> K 0..7,
// h8..h15 -> K 16..23; lanes 16-31 same rows, K shifted by +8.
// LDS tile is row-major [rows][ld] (ld in halves, row stride multiple of 8).
__device__ __forceinline__ v16h frag_a(const _Float16* lds, int ld, int row,
                                       int k0, int lane) {
    int kb = k0 + ((lane >> 4) << 3);
    const _Float16* p = lds + row * ld;
    v8h lo = *(const v8h*)(p + kb);
    v8h hi = *(const v8h*)(p + kb + 16);
    return make_v16(lo, hi);
}

// B fragment (32x16 f16, K x N) from a TRANSPOSED LDS tile Bt[n][k]:
// lanes 0-15 = cols (N), h0..h7 -> K 0..7, h8..h15 -> K 16..23; lanes 16-31 same cols, K +8.
__device__ __forceinline__ v16h frag_b(const _Float16* lds, int ld, int col,
                                       int k0, int lane) {
    int kb = k0 + ((lane >> 4) << 3);
    const _Float16* p = lds + col * ld + kb;
    v8h lo = *(const v8h*)(p);
    v8h hi = *(const v8h*)(p + 16);
    return make_v16(lo, hi);
}

// ---------------------------------------------------------------------------
// 128x64-tile, double-buffered WMMA GEMM: C = A[M,K] @ op(B) (+epilogues)
//   8 waves, each owns a 2x2 grid of 16x16 subtiles: 4 WMMAs / 8 frag loads
//   per 32-wide k-step; LDS double buffering overlaps staging with WMMA.
//   TRANSB=1 : B' is [N,K] row-major, op(B)[k][n] = B'[n][k]  (direct staging)
//   MODE 0: out32[m*ldo32+n] = acc            (only n < Nout)     [scores]
//   MODE 1: s = acc + addsrc[m*768+n]; out32 = s; out16 = f16(s)  [h_truth]
//   MODE 2: out16 = f16(acc + bias[n])                            [Q/K/V]
//   MODE 3: out32 = acc + bias[n]                                 [Wo]
// ---------------------------------------------------------------------------
template <bool TRANSB, int MODE>
__global__ __launch_bounds__(256) void gemm128(
    const _Float16* __restrict__ A, int lda,
    const _Float16* __restrict__ B, int ldb,
    const float* __restrict__ bias,
    const float* __restrict__ addsrc,
    float* __restrict__ out32, int ldo32,
    _Float16* __restrict__ out16,
    int K, int Nout) {
    __shared__ _Float16 sA[2][128 * 40];  // 2 x (128 rows x 32 K), stride 40
    __shared__ _Float16 sB[2][64 * 40];   // 2 x (64 N x 32 K), transposed

    const int tid = threadIdx.x;
    const int lane = tid & 31, wave = tid >> 5;
    const int m0 = blockIdx.y * 128, n0 = blockIdx.x * 64;
    const int ms0 = (wave & 3) * 2;   // two m-subtiles: ms0, ms0+1
    const int ns0 = (wave >> 2) * 2;  // two n-subtiles: ns0, ns0+1

    auto stage = [&](int buf, int k0) {
        // A: 128x32, two b128 per thread
#pragma unroll
        for (int j = 0; j < 2; ++j) {
            int row = tid >> 1, cb = (tid & 1) * 16 + j * 8;
            *(v8h*)&sA[buf][row * 40 + cb] =
                *(const v8h*)&A[(size_t)(m0 + row) * lda + k0 + cb];
            if (j == 0)  // warm L2/WGP$ two k-tiles ahead (speculative, OOB-safe)
                __builtin_prefetch(&A[(size_t)(m0 + row) * lda + k0 + 64 + cb], 0, 3);
        }
        if (TRANSB) {  // B'[n][k] row-major -> direct copy into Bt[n][k]
            int row = tid >> 2, cb = (tid & 3) * 8;
            *(v8h*)&sB[buf][row * 40 + cb] =
                *(const v8h*)&B[(size_t)(n0 + row) * ldb + k0 + cb];
        } else {       // B[k][n] row-major -> scatter into Bt[n][k]
            int k = tid >> 3, nb = (tid & 7) * 8;
            v8h v = *(const v8h*)&B[(size_t)(k0 + k) * ldb + n0 + nb];
#pragma unroll
            for (int i = 0; i < 8; ++i) sB[buf][(nb + i) * 40 + k] = v[i];
        }
    };

    v8f acc00 = {}, acc01 = {}, acc10 = {}, acc11 = {};
    stage(0, 0);
    const int nk = K / 32;
    for (int ki = 0; ki < nk; ++ki) {
        __syncthreads();
        if (ki + 1 < nk) stage((ki + 1) & 1, (ki + 1) * 32);
        const _Float16* a_ = sA[ki & 1];
        const _Float16* b_ = sB[ki & 1];
        v16h a0 = frag_a(a_, 40, (ms0 + 0) * 16 + (lane & 15), 0, lane);
        v16h a1 = frag_a(a_, 40, (ms0 + 1) * 16 + (lane & 15), 0, lane);
        v16h b0 = frag_b(b_, 40, (ns0 + 0) * 16 + (lane & 15), 0, lane);
        v16h b1 = frag_b(b_, 40, (ns0 + 1) * 16 + (lane & 15), 0, lane);
        acc00 = wmma_f16(a0, b0, acc00);
        acc01 = wmma_f16(a0, b1, acc01);
        acc10 = wmma_f16(a1, b0, acc10);
        acc11 = wmma_f16(a1, b1, acc11);
    }

    // Epilogue.  C/D layout: lanes 0-15 -> N=lane, M rows r; lanes 16-31 -> M+8.
    const int nloc = lane & 15;
#pragma unroll
    for (int mi = 0; mi < 2; ++mi) {
        int mbase = m0 + (ms0 + mi) * 16 + ((lane >> 4) << 3);
#pragma unroll
        for (int ni = 0; ni < 2; ++ni) {
            v8f acc = mi == 0 ? (ni == 0 ? acc00 : acc01)
                              : (ni == 0 ? acc10 : acc11);
            int n = n0 + (ns0 + ni) * 16 + nloc;
            float bv = (MODE == 2 || MODE == 3) ? bias[n] : 0.0f;
#pragma unroll
            for (int r = 0; r < 8; ++r) {
                int m = mbase + r;
                float v = acc[r] + bv;
                if (MODE == 0) {
                    if (n < Nout) out32[(size_t)m * ldo32 + n] = v;
                } else if (MODE == 1) {
                    float s = v + addsrc[(size_t)m * EMB + n];
                    out32[(size_t)m * EMB + n] = s;
                    out16[(size_t)m * EMB + n] = (_Float16)s;
                } else if (MODE == 2) {
                    out16[(size_t)m * EMB + n] = (_Float16)v;
                } else {
                    out32[(size_t)m * EMB + n] = v;
                }
            }
        }
    }
}

// ---------------------------------------------------------------------------
// Fused two-stream masked attention: per (b, h, 16-row q-tile)
//   logits = Q·K^T / 8 + (1-mask)*NEG + diag*NEG ; softmax ; ctx = P·V
// ---------------------------------------------------------------------------
__global__ __launch_bounds__(128) void attn_kernel(
    const _Float16* __restrict__ Qh, const _Float16* __restrict__ Kh,
    const _Float16* __restrict__ Vh, const float* __restrict__ mask,
    _Float16* __restrict__ ctx) {
    __shared__ _Float16 sQ[16 * 72];    //  2.25 KB : Q tile [16 x 64]
    __shared__ _Float16 sB[64 * 72];    //  9.0 KB : K chunk (as Bt) / V^T chunk
    __shared__ float    sS[16 * 520];   // 33.3 KB : logits [16 x 512]
    __shared__ _Float16 sP[16 * 520];   // 16.6 KB : probs  [16 x 512] f16

    const int tid = threadIdx.x, lane = tid & 31, wave = tid >> 5;
    const int qt = blockIdx.x, h = blockIdx.y, b = blockIdx.z;
    const int q0 = qt * 16;
    const size_t rowb = (size_t)b * SEQ;

    {   // stage Q [16 x 64]
        int r = tid >> 3, db = (tid & 7) * 8;
        *(v8h*)&sQ[r * 72 + db] =
            *(const v8h*)&Qh[(rowb + q0 + r) * EMB + h * HDIM + db];
    }

    // ---- phase 1: logits over all 512 keys, 64 at a time -------------------
    for (int nb = 0; nb < 8; ++nb) {
        __syncthreads();
        {   // stage K chunk [64 kv x 64 d] == Bt[n=kv][k=d] directly
            int r = tid >> 1;
#pragma unroll
            for (int j = 0; j < 4; ++j) {
                int db = (tid & 1) * 32 + j * 8;
                *(v8h*)&sB[r * 72 + db] =
                    *(const v8h*)&Kh[(rowb + nb * 64 + r) * EMB + h * HDIM + db];
            }
        }
        __syncthreads();

        v8f acc = {};
#pragma unroll
        for (int ks = 0; ks < 2; ++ks) {
            v16h a  = frag_a(sQ, 72, lane & 15, ks * 32, lane);
            v16h bb = frag_b(sB, 72, wave * 16 + (lane & 15), ks * 32, lane);
            acc = wmma_f16(a, bb, acc);
        }
        int nloc = lane & 15;
        int kv = nb * 64 + wave * 16 + nloc;
        float pb = (1.0f - mask[rowb + kv]) * NEGB;
        int mb = (lane >> 4) << 3;
#pragma unroll
        for (int r = 0; r < 8; ++r) {
            int m = mb + r;
            float v = acc[r] * 0.125f + pb + ((q0 + m) == kv ? NEGB : 0.0f);
            sS[m * 520 + kv] = v;
        }
    }
    __syncthreads();

    // ---- phase 2: row softmax (8 lanes per row, 64 cols each) --------------
    {
        int r = tid >> 3, cb = (tid & 7) * 64;
        float vbuf[64];
        float mx = -3.4e38f;
#pragma unroll
        for (int j = 0; j < 64; ++j) {
            vbuf[j] = sS[r * 520 + cb + j];
            mx = fmaxf(mx, vbuf[j]);
        }
        for (int o = 1; o < 8; o <<= 1) mx = fmaxf(mx, __shfl_xor(mx, o, 32));
        float sum = 0.0f;
#pragma unroll
        for (int j = 0; j < 64; ++j) { vbuf[j] = __expf(vbuf[j] - mx); sum += vbuf[j]; }
        for (int o = 1; o < 8; o <<= 1) sum += __shfl_xor(sum, o, 32);
        float inv = 1.0f / sum;
#pragma unroll
        for (int j = 0; j < 64; ++j) sP[r * 520 + cb + j] = (_Float16)(vbuf[j] * inv);
    }

    // ---- phase 3: ctx = P[16x512] @ V[512x64] ------------------------------
    v8f acc = {};
    for (int ch = 0; ch < 8; ++ch) {
        __syncthreads();
        {   // stage V chunk transposed: Bt[n=d][k=kv]
            int kv = tid >> 1;
#pragma unroll
            for (int j = 0; j < 4; ++j) {
                int db = (tid & 1) * 32 + j * 8;
                v8h v = *(const v8h*)&Vh[(rowb + ch * 64 + kv) * EMB + h * HDIM + db];
#pragma unroll
                for (int i = 0; i < 8; ++i) sB[(db + i) * 72 + kv] = v[i];
            }
        }
        __syncthreads();
#pragma unroll
        for (int ks = 0; ks < 2; ++ks) {
            v16h a  = frag_a(sP, 520, lane & 15, ch * 64 + ks * 32, lane);
            v16h bb = frag_b(sB, 72, wave * 16 + (lane & 15), ks * 32, lane);
            acc = wmma_f16(a, bb, acc);
        }
    }
    {
        int d = wave * 16 + (lane & 15);
        int mb = (lane >> 4) << 3;
#pragma unroll
        for (int r = 0; r < 8; ++r) {
            int m = mb + r;
            ctx[(rowb + q0 + m) * EMB + h * HDIM + d] = (_Float16)acc[r];
        }
    }
}

// ---------------------------------------------------------------------------
// Label softmax: rows of 100 f32 -> f16 probs padded to 128 cols (zeros)
// ---------------------------------------------------------------------------
__global__ __launch_bounds__(256) void label_softmax_kernel(
    const float* __restrict__ scores, _Float16* __restrict__ atth) {
    int row = blockIdx.x * 8 + (threadIdx.x >> 5);
    int lane = threadIdx.x & 31;
    float v[4];
    float mx = -3.4e38f;
#pragma unroll
    for (int j = 0; j < 4; ++j) {
        int c = lane + j * 32;
        v[j] = (c < NLBL) ? scores[(size_t)row * NLBL + c] : -3.4e38f;
        mx = fmaxf(mx, v[j]);
    }
    for (int o = 1; o < 32; o <<= 1) mx = fmaxf(mx, __shfl_xor(mx, o, 32));
    float sum = 0.0f;
#pragma unroll
    for (int j = 0; j < 4; ++j) {
        int c = lane + j * 32;
        v[j] = (c < NLBL) ? __expf(v[j] - mx) : 0.0f;
        sum += v[j];
    }
    for (int o = 1; o < 32; o <<= 1) sum += __shfl_xor(sum, o, 32);
    float inv = 1.0f / sum;
#pragma unroll
    for (int j = 0; j < 4; ++j) {
        int c = lane + j * 32;
        atth[(size_t)row * LPAD + c] = (_Float16)(v[j] * inv);
    }
}

// ---------------------------------------------------------------------------
// Final fused epilogue: out = LN(attn_out + X + tp)*g + b + h_truth
// One wave per row (768 = 24 elems/lane).
// ---------------------------------------------------------------------------
__global__ __launch_bounds__(256) void ln_out_kernel(
    const float* __restrict__ ao, const float* __restrict__ X,
    const float* __restrict__ tp, const float* __restrict__ g,
    const float* __restrict__ bta, const float* __restrict__ ht32,
    float* __restrict__ out) {
    int row = blockIdx.x * 8 + (threadIdx.x >> 5);
    int lane = threadIdx.x & 31;
    float y[24];
    float s = 0.0f;
#pragma unroll
    for (int j = 0; j < 24; ++j) {
        int c = lane + j * 32;
        y[j] = ao[(size_t)row * EMB + c] + X[(size_t)row * EMB + c] + tp[c];
        s += y[j];
    }
    for (int o = 1; o < 32; o <<= 1) s += __shfl_xor(s, o, 32);
    float mu = s * (1.0f / EMB);
    float s2 = 0.0f;
#pragma unroll
    for (int j = 0; j < 24; ++j) { float d = y[j] - mu; s2 += d * d; }
    for (int o = 1; o < 32; o <<= 1) s2 += __shfl_xor(s2, o, 32);
    float rinv = rsqrtf(s2 * (1.0f / EMB) + 1e-12f);
#pragma unroll
    for (int j = 0; j < 24; ++j) {
        int c = lane + j * 32;
        out[(size_t)row * EMB + c] =
            (y[j] - mu) * rinv * g[c] + bta[c] + ht32[(size_t)row * EMB + c];
    }
}

// ---------------------------------------------------------------------------
// Prep / conversion kernels
// ---------------------------------------------------------------------------
__global__ __launch_bounds__(256) void prep_x_kernel(
    const float* __restrict__ X, const float* __restrict__ tp,
    _Float16* __restrict__ Xh, _Float16* __restrict__ Xuh) {
    size_t i = (size_t)blockIdx.x * 256 + threadIdx.x;
    int c = (int)(i % EMB);
    float x = X[i];
    Xh[i]  = (_Float16)x;
    Xuh[i] = (_Float16)(x + tp[c]);
}

__global__ __launch_bounds__(256) void prep_t_kernel(
    const float* __restrict__ T, _Float16* __restrict__ Th) {
    size_t i = (size_t)blockIdx.x * 256 + threadIdx.x;
    int l = (int)(i / EMB);
    Th[i] = (l < NLBL) ? (_Float16)T[i] : (_Float16)0.0f;
}

__global__ __launch_bounds__(256) void cvt_kernel(
    const float* __restrict__ src, _Float16* __restrict__ dst) {
    size_t i = (size_t)blockIdx.x * 256 + threadIdx.x;
    dst[i] = (_Float16)src[i];
}

// ---------------------------------------------------------------------------
// Host-side launch
// ---------------------------------------------------------------------------
extern "C" void kernel_launch(void* const* d_in, const int* in_sizes, int n_in,
                              void* d_out, int out_size, void* d_ws,
                              size_t ws_size, hipStream_t stream) {
    (void)in_sizes; (void)n_in; (void)out_size; (void)ws_size;

    const float* X    = (const float*)d_in[0];
    const float* mask = (const float*)d_in[1];
    const float* T    = (const float*)d_in[2];
    const float* tp   = (const float*)d_in[3];
    const float* Wq   = (const float*)d_in[4];
    const float* bq   = (const float*)d_in[5];
    const float* Wk   = (const float*)d_in[6];
    const float* bk   = (const float*)d_in[7];
    const float* Wv   = (const float*)d_in[8];
    const float* bv   = (const float*)d_in[9];
    const float* Wo   = (const float*)d_in[10];
    const float* bo   = (const float*)d_in[11];
    const float* lng  = (const float*)d_in[12];
    const float* lnb  = (const float*)d_in[13];

    float* out    = (float*)d_out;                       // [16,512,768]
    float* scores = out + (size_t)ROWS * EMB;            // [16,512,100]

    // workspace carve-out
    char* w = (char*)d_ws;
    auto alloc = [&](size_t bytes) -> void* {
        void* p = (void*)w;
        w += (bytes + 255) & ~(size_t)255;
        return p;
    };
    const size_t R = (size_t)ROWS * EMB;
    _Float16* Xh   = (_Float16*)alloc(R * 2);
    _Float16* Xuh  = (_Float16*)alloc(R * 2);
    _Float16* Th   = (_Float16*)alloc((size_t)LPAD * EMB * 2);
    _Float16* atth = (_Float16*)alloc((size_t)ROWS * LPAD * 2);
    _Float16* ht16 = (_Float16*)alloc(R * 2);
    _Float16* Qh   = (_Float16*)alloc(R * 2);
    _Float16* Kh   = (_Float16*)alloc(R * 2);
    _Float16* Vh   = (_Float16*)alloc(R * 2);
    _Float16* ctxh = (_Float16*)alloc(R * 2);
    _Float16* Wqh  = (_Float16*)alloc((size_t)EMB * EMB * 2);
    _Float16* Wkh  = (_Float16*)alloc((size_t)EMB * EMB * 2);
    _Float16* Wvh  = (_Float16*)alloc((size_t)EMB * EMB * 2);
    _Float16* Woh  = (_Float16*)alloc((size_t)EMB * EMB * 2);
    float*    ht32 = (float*)alloc(R * 4);
    float*    ao   = (float*)alloc(R * 4);

    const dim3 blk256(256);

    // 0) conversions
    prep_x_kernel<<<(unsigned)(R / 256), blk256, 0, stream>>>(X, tp, Xh, Xuh);
    prep_t_kernel<<<(LPAD * EMB) / 256, blk256, 0, stream>>>(T, Th);
    cvt_kernel<<<(EMB * EMB) / 256, blk256, 0, stream>>>(Wq, Wqh);
    cvt_kernel<<<(EMB * EMB) / 256, blk256, 0, stream>>>(Wk, Wkh);
    cvt_kernel<<<(EMB * EMB) / 256, blk256, 0, stream>>>(Wv, Wvh);
    cvt_kernel<<<(EMB * EMB) / 256, blk256, 0, stream>>>(Wo, Woh);

    // 1) scores = X @ T^T  -> d_out scores region (f32, ld=100, cols<100)
    gemm128<true, 0><<<dim3(LPAD / 64, ROWS / 128), blk256, 0, stream>>>(
        Xh, EMB, Th, EMB, nullptr, nullptr, scores, NLBL, nullptr, EMB, NLBL);

    // 2) att = softmax(scores) -> f16 padded [8192,128]
    label_softmax_kernel<<<ROWS / 8, blk256, 0, stream>>>(scores, atth);

    // 3) h_truth = X + att @ T  (f32 + f16)
    gemm128<false, 1><<<dim3(EMB / 64, ROWS / 128), blk256, 0, stream>>>(
        atth, LPAD, Th, EMB, nullptr, X, ht32, EMB, ht16, LPAD, EMB);

    // 4) Q = (X+tp) @ Wq + bq ; K/V from h_truth
    gemm128<false, 2><<<dim3(EMB / 64, ROWS / 128), blk256, 0, stream>>>(
        Xuh, EMB, Wqh, EMB, bq, nullptr, nullptr, 0, Qh, EMB, EMB);
    gemm128<false, 2><<<dim3(EMB / 64, ROWS / 128), blk256, 0, stream>>>(
        ht16, EMB, Wkh, EMB, bk, nullptr, nullptr, 0, Kh, EMB, EMB);
    gemm128<false, 2><<<dim3(EMB / 64, ROWS / 128), blk256, 0, stream>>>(
        ht16, EMB, Wvh, EMB, bv, nullptr, nullptr, 0, Vh, EMB, EMB);

    // 5) fused masked attention
    attn_kernel<<<dim3(SEQ / 16, HEADS, BATCH), dim3(128), 0, stream>>>(
        Qh, Kh, Vh, mask, ctxh);

    // 6) attn_out = ctx @ Wo + bo (f32)
    gemm128<false, 3><<<dim3(EMB / 64, ROWS / 128), blk256, 0, stream>>>(
        ctxh, EMB, Woh, EMB, bo, nullptr, ao, EMB, nullptr, EMB, EMB);

    // 7) out = LN(attn_out + X + tp)*g + b + h_truth
    ln_out_kernel<<<ROWS / 8, blk256, 0, stream>>>(ao, X, tp, lng, lnb, ht32, out);
}
